// LAttn_57423712747928
// MI455X (gfx1250) — compile-verified, matrix operand order checked by reading, not run on hardware
//
#include <hip/hip_runtime.h>

// ---------------- problem constants (match reference) ----------------
constexpr int NG = 64;      // groups
constexpr int SP = 1024;    // points per group (S)
constexpr int FD = 512;     // feature dim (D)
constexpr float EPS_INV = 20.0f;  // 1/0.05

typedef __attribute__((ext_vector_type(4)))  _Float16 v4h;
typedef __attribute__((ext_vector_type(8)))  _Float16 v8h;
typedef __attribute__((ext_vector_type(16))) _Float16 v16h;
typedef __attribute__((ext_vector_type(8)))  __bf16   v8bf;
typedef __attribute__((ext_vector_type(16))) __bf16   v16bf;
typedef __attribute__((ext_vector_type(8)))  float    v8f;

static __device__ __forceinline__ float wave_reduce_sum(float x) {
  #pragma unroll
  for (int off = 16; off > 0; off >>= 1) x += __shfl_xor(x, off, 32);
  return x;
}

// Build a 16-half fragment per the CDNA5 16-bit A/B layout:
// lane holds row (lane&15); K chunk base kh=(lane>>4)*8; halves [kh..kh+7] and [kh+16..kh+23].
static __device__ __forceinline__ v16h load_frag_f16(const _Float16* p) {
  v8h lo = *(const v8h*)p;
  v8h hi = *(const v8h*)(p + 16);
  return __builtin_shufflevector(lo, hi, 0,1,2,3,4,5,6,7,8,9,10,11,12,13,14,15);
}
static __device__ __forceinline__ v16bf load_frag_bf16(const __bf16* p) {
  v8bf lo = *(const v8bf*)p;
  v8bf hi = *(const v8bf*)(p + 16);
  return __builtin_shufflevector(lo, hi, 0,1,2,3,4,5,6,7,8,9,10,11,12,13,14,15);
}

// ---------------- 1) row L2-normalize -> vn (f16) ----------------
// one wave per row of 512 f32
__global__ void __launch_bounds__(256) normalize_kernel(const float* __restrict__ v,
                                                        _Float16* __restrict__ vn) {
  const int row  = (int)((blockIdx.x * blockDim.x + threadIdx.x) >> 5);
  const int lane = threadIdx.x & 31;
  const float4* rp = (const float4*)(v + (size_t)row * FD);
  float4 x[4];
  float ss = 0.0f;
  #pragma unroll
  for (int t = 0; t < 4; ++t) {
    x[t] = rp[t * 32 + lane];
    ss += x[t].x * x[t].x + x[t].y * x[t].y + x[t].z * x[t].z + x[t].w * x[t].w;
  }
  ss = wave_reduce_sum(ss);
  const float inv = 1.0f / fmaxf(sqrtf(ss), 1e-12f);
  v4h* op = (v4h*)(vn + (size_t)row * FD);
  #pragma unroll
  for (int t = 0; t < 4; ++t) {
    v4h h;
    h[0] = (_Float16)(x[t].x * inv);
    h[1] = (_Float16)(x[t].y * inv);
    h[2] = (_Float16)(x[t].z * inv);
    h[3] = (_Float16)(x[t].w * inv);
    op[t * 32 + lane] = h;
  }
}

// ---------------- 2) transpose v -> vT bf16  [G][D][S] ----------------
__global__ void __launch_bounds__(256) transpose_kernel(const float* __restrict__ v,
                                                        __bf16* __restrict__ vT) {
  __shared__ __bf16 tile[64][65];
  const int bx = blockIdx.x;
  const int g  = bx >> 7;
  const int t  = bx & 127;
  const int s0 = (t >> 3) << 6;
  const int d0 = (t & 7) << 6;
  const int tid = threadIdx.x;
  const int di  = tid & 63;
  const int si0 = tid >> 6;       // 0..3
  #pragma unroll
  for (int k = 0; k < 16; ++k) {
    const int si = si0 + 4 * k;
    const float x = v[(size_t)(g * SP + s0 + si) * FD + d0 + di];
    tile[si][di] = (__bf16)x;
  }
  __syncthreads();
  const int sj  = tid & 63;
  const int dj0 = tid >> 6;
  #pragma unroll
  for (int k = 0; k < 16; ++k) {
    const int dj = dj0 + 4 * k;
    vT[(size_t)(g * FD + d0 + dj) * SP + s0 + sj] = tile[sj][dj];
  }
}

// ---------------- 3) GEMM1: K = exp(20*(vn vn^T - 1))  (f16 WMMA) ----------------
// 128x128 tile per block, 8 waves each 64x32 (4 M-tiles x 2 N-tiles)
__global__ void __launch_bounds__(256) gemm1_exp_kernel(const _Float16* __restrict__ vn,
                                                        __bf16* __restrict__ Kb) {
  const int bx = blockIdx.x;
  const int g  = bx >> 6;
  const int t  = bx & 63;
  const int tm = (t >> 3) << 7;
  const int tn = (t & 7) << 7;
  const int w    = threadIdx.x >> 5;
  const int lane = threadIdx.x & 31;
  const int wm  = tm + ((w >> 2) << 6);
  const int wn  = tn + ((w & 3) << 5);
  const int r16 = lane & 15;
  const int kh  = (lane >> 4) << 3;

  v8f acc[4][2];
  #pragma unroll
  for (int i = 0; i < 4; ++i)
    #pragma unroll
    for (int j = 0; j < 2; ++j) { v8f z = {}; acc[i][j] = z; }

  const _Float16* base = vn + (size_t)g * SP * FD;
  for (int k0 = 0; k0 < FD; k0 += 32) {
    v16h af[4], bfr[2];
    #pragma unroll
    for (int i = 0; i < 4; ++i)
      af[i] = load_frag_f16(base + (size_t)(wm + 16 * i + r16) * FD + k0 + kh);
    #pragma unroll
    for (int j = 0; j < 2; ++j)
      bfr[j] = load_frag_f16(base + (size_t)(wn + 16 * j + r16) * FD + k0 + kh);
    #pragma unroll
    for (int i = 0; i < 4; ++i)
      #pragma unroll
      for (int j = 0; j < 2; ++j)
        acc[i][j] = __builtin_amdgcn_wmma_f32_16x16x32_f16(
            false, af[i], false, bfr[j], (short)0, acc[i][j], false, false);
  }

  const int mh = (lane >> 4) << 3;
  __bf16* kout = Kb + (size_t)g * SP * SP;
  #pragma unroll
  for (int i = 0; i < 4; ++i)
    #pragma unroll
    for (int j = 0; j < 2; ++j)
      #pragma unroll
      for (int r = 0; r < 8; ++r) {
        const int m = wm + 16 * i + mh + r;
        const int n = wn + 16 * j + r16;
        const float Tv = __expf(EPS_INV * (acc[i][j][r] - 1.0f));
        kout[(size_t)m * SP + n] = (__bf16)Tv;
      }
}

// ---------------- 4) Sinkhorn scaling-vector matvecs ----------------
__global__ void __launch_bounds__(256) init_c_kernel(float* __restrict__ c) {
  c[blockIdx.x * 256 + threadIdx.x] = 1.0f;
}

// dst[i] = 1 / sum_j K[i,j]*c[j]   (one wave per row)
__global__ void __launch_bounds__(256) row_mv_kernel(const __bf16* __restrict__ Kb,
                                                     const float* __restrict__ c,
                                                     float* __restrict__ dst) {
  const int row  = (int)((blockIdx.x * blockDim.x + threadIdx.x) >> 5); // 0..NG*SP-1
  const int lane = threadIdx.x & 31;
  const int g    = row >> 10;
  const __bf16* kp = Kb + (size_t)row * SP;
  const float*  cg = c + (g << 10);
  float sum = 0.0f;
  #pragma unroll
  for (int t = 0; t < 4; ++t) {
    const int j0 = t * 256 + lane * 8;
    v8bf kv = *(const v8bf*)(kp + j0);
    #pragma unroll
    for (int e = 0; e < 8; ++e) sum += (float)kv[e] * cg[j0 + e];
  }
  sum = wave_reduce_sum(sum);
  if (lane == 0) dst[row] = 1.0f / sum;
}

// c[j] = 1 / sum_i K[i,j]*r[i]   (one thread per column, coalesced across lanes)
__global__ void __launch_bounds__(256) col_mv_kernel(const __bf16* __restrict__ Kb,
                                                     const float* __restrict__ r,
                                                     float* __restrict__ c) {
  const int idx = blockIdx.x * 256 + threadIdx.x;   // 0..NG*SP-1
  const int g   = idx >> 10;
  const int j   = idx & 1023;
  const __bf16* kp = Kb + ((size_t)g << 20) + j;
  const float*  rg = r + (g << 10);
  float sum = 0.0f;
  for (int i = 0; i < SP; i += 4) {
    sum += (float)kp[(size_t)(i + 0) * SP] * rg[i + 0];
    sum += (float)kp[(size_t)(i + 1) * SP] * rg[i + 1];
    sum += (float)kp[(size_t)(i + 2) * SP] * rg[i + 2];
    sum += (float)kp[(size_t)(i + 3) * SP] * rg[i + 3];
  }
  c[idx] = 1.0f / sum;
}

// ---------------- 5) fold c into vT:  vT[g][d][s] *= c[g][s] ----------------
__global__ void __launch_bounds__(256) scale_vt_kernel(__bf16* __restrict__ vT,
                                                       const float* __restrict__ c) {
  const size_t base = ((size_t)blockIdx.x * 256 + threadIdx.x) * 8;
  #pragma unroll
  for (int e = 0; e < 8; ++e) {
    const size_t f = base + e;
    const int s = (int)(f & 1023);
    const int g = (int)(f >> 19);           // D*S = 2^19
    vT[f] = (__bf16)((float)vT[f] * c[(g << 10) + s]);
  }
}

// ---------------- 6) GEMM2: out = diag(a) K_c vT_c   (bf16 WMMA) ----------------
__global__ void __launch_bounds__(256) gemm2_kernel(const __bf16* __restrict__ Kb,
                                                    const __bf16* __restrict__ vTc,
                                                    const float* __restrict__ avec,
                                                    float* __restrict__ out) {
  const int bx = blockIdx.x;
  const int g  = bx >> 5;
  const int t  = bx & 31;
  const int tm = (t >> 2) << 7;  // 8 M-tiles over S
  const int tn = (t & 3) << 7;   // 4 N-tiles over D
  const int w    = threadIdx.x >> 5;
  const int lane = threadIdx.x & 31;
  const int wm  = tm + ((w >> 2) << 6);
  const int wn  = tn + ((w & 3) << 5);
  const int r16 = lane & 15;
  const int kh  = (lane >> 4) << 3;

  v8f acc[4][2];
  #pragma unroll
  for (int i = 0; i < 4; ++i)
    #pragma unroll
    for (int j = 0; j < 2; ++j) { v8f z = {}; acc[i][j] = z; }

  const __bf16* ka = Kb  + (size_t)g * SP * SP;
  const __bf16* vb = vTc + (size_t)g * FD * SP;
  for (int k0 = 0; k0 < SP; k0 += 32) {
    v16bf af[4], bfr[2];
    #pragma unroll
    for (int i = 0; i < 4; ++i)
      af[i] = load_frag_bf16(ka + (size_t)(wm + 16 * i + r16) * SP + k0 + kh);
    #pragma unroll
    for (int j = 0; j < 2; ++j)
      bfr[j] = load_frag_bf16(vb + (size_t)(wn + 16 * j + r16) * SP + k0 + kh);
    #pragma unroll
    for (int i = 0; i < 4; ++i)
      #pragma unroll
      for (int j = 0; j < 2; ++j)
        acc[i][j] = __builtin_amdgcn_wmma_f32_16x16x32_bf16(
            false, af[i], false, bfr[j], (short)0, acc[i][j], false, false);
  }

  const int mh = (lane >> 4) << 3;
  const float* ag = avec + (g << 10);
  float* og = out + (size_t)g * SP * FD;
  #pragma unroll
  for (int i = 0; i < 4; ++i)
    #pragma unroll
    for (int j = 0; j < 2; ++j)
      #pragma unroll
      for (int r = 0; r < 8; ++r) {
        const int m = wm + 16 * i + mh + r;
        const int n = wn + 16 * j + r16;
        og[(size_t)m * FD + n] = acc[i][j][r] * ag[m];
      }
}

// ---------------- host: launch pipeline ----------------
extern "C" void kernel_launch(void* const* d_in, const int* in_sizes, int n_in,
                              void* d_out, int out_size, void* d_ws, size_t ws_size,
                              hipStream_t stream) {
  (void)in_sizes; (void)n_in; (void)out_size; (void)ws_size;
  const float* v = (const float*)d_in[0];   // v_feats [N, D] f32 (others unused)

  char* ws = (char*)d_ws;
  _Float16* vn  = (_Float16*)ws;                              //  64 MB
  __bf16*   vT  = (__bf16*)(ws + (size_t)67108864);           //  64 MB
  __bf16*   Kb  = (__bf16*)(ws + (size_t)134217728);          // 128 MB
  float*    cvec = (float*)(ws + (size_t)268435456);          // 256 KB
  float*    rvec = cvec + NG * SP;
  float*    avec = rvec + NG * SP;
  float*    out  = (float*)d_out;

  normalize_kernel<<<8192, 256, 0, stream>>>(v, vn);          // 65536 rows, wave/row
  transpose_kernel<<<8192, 256, 0, stream>>>(v, vT);          // 64x64 tiles
  gemm1_exp_kernel<<<NG * 64, 256, 0, stream>>>(vn, Kb);      // 128x128 tiles
  init_c_kernel<<<256, 256, 0, stream>>>(cvec);
  for (int it = 0; it < 3; ++it) {
    row_mv_kernel<<<8192, 256, 0, stream>>>(Kb, cvec, rvec);
    col_mv_kernel<<<256, 256, 0, stream>>>(Kb, rvec, cvec);
  }
  row_mv_kernel<<<8192, 256, 0, stream>>>(Kb, cvec, avec);    // final row scale a
  scale_vt_kernel<<<16384, 256, 0, stream>>>(vT, cvec);       // fold c into B
  gemm2_kernel<<<NG * 32, 256, 0, stream>>>(Kb, vT, avec, out);
}
